// Model_7413113553118
// MI455X (gfx1250) — matrix operand order repaired and checked by reading, not compiled, not run on hardware
//
#include <hip/hip_runtime.h>
#include <hip/hip_bf16.h>

// ---------------- problem constants (from reference) ----------------
#define B_    32
#define S_    24
#define NN_   103
#define NF_   2
#define K_    2
#define LBL_  24
#define H1_   16
#define H2_   8
#define E_    1648
#define INP_  7410
#define DV_   1852
#define DD_   1860          // d_model
#define NH_   10
#define HD_   186           // head dim
#define DFF_  512
#define NL_   4
#define TOK_  (B_*NN_)      // 3296
#define FLAT_ (S_*NN_)      // 2472

typedef __attribute__((ext_vector_type(16))) _Float16 v16h;
typedef __attribute__((ext_vector_type(8)))  _Float16 v8h;
typedef __attribute__((ext_vector_type(8)))  float    v8f;

union V16 { v16h v; v8h h[2]; };

// =====================================================================
// Generic tiled WMMA GEMM:  C[M,N] = act( A[M,K] * Bm[K,N] + bias + res )
// A,Bm,res,C row-major f32; converted to f16 in LDS; f32 accumulate.
// 256 threads = 8 waves; block tile 128x128; wave grid 4x2; wave tile
// 32x64 = 2x4 WMMA fragments -> 8 v_wmma per K-step per wave.
// B tile staged TRANSPOSED (Bst[n][k]) so each lane's fragment is two
// contiguous 16-byte LDS reads (ds_load_b128), same as A.
// =====================================================================
#define BM 128
#define BN 128
#define BK 32

__global__ __launch_bounds__(256) void gemm_wmma_f16(
    const float* __restrict__ A, const float* __restrict__ Bm,
    const float* __restrict__ bias, const float* __restrict__ res,
    float* __restrict__ C, int M, int N, int K, int relu)
{
    __shared__ _Float16 As[BM][BK];    // 8 KB, row-major
    __shared__ _Float16 Bst[BN][BK];   // 8 KB, transposed (n-major)

    const int tid  = threadIdx.x;
    const int lane = tid & 31;
    const int wave = tid >> 5;
    const int bm0  = blockIdx.y * BM;
    const int bn0  = blockIdx.x * BN;
    const int wm0  = (wave >> 1) * 32;   // 0,32,64,96
    const int wn0  = (wave & 1) * 64;    // 0 or 64
    const int half = lane >> 4;
    const int l16  = lane & 15;

    const bool mfull = (bm0 + BM <= M);
    const bool nfull = (bn0 + BN <= N);

    v8f acc[2][4] = {};

    for (int k0 = 0; k0 < K; k0 += BK) {
        const bool kfull = (k0 + BK <= K);

        // ---------------- stage A tile (128x32) ----------------
        if (kfull && mfull) {
            // two row-chunks of 8 f32 per thread; K is even for all GEMMs
            // here, so float2 (8B) accesses stay aligned on every row.
            #pragma unroll
            for (int it = 0; it < 2; ++it) {
                int idx = tid + it * 256;
                int r = idx >> 2;
                int c = (idx & 3) << 3;
                const float* src = A + (long)(bm0 + r) * K + (k0 + c);
                v8h pk;
                #pragma unroll
                for (int j = 0; j < 8; j += 2) {
                    float2 f = *(const float2*)(src + j);
                    pk[j]     = (_Float16)f.x;
                    pk[j + 1] = (_Float16)f.y;
                }
                *(v8h*)&As[r][c] = pk;
            }
        } else {
            for (int i = tid; i < BM * BK; i += 256) {
                int r = i >> 5, c = i & 31;
                int gr = bm0 + r, gc = k0 + c;
                As[r][c] = (_Float16)((gr < M && gc < K) ? A[(long)gr * K + gc] : 0.0f);
            }
        }

        // ---------------- stage B tile (32x128), transposed ----------------
        if (kfull && nfull) {
            // each thread owns one column (n) for 16 of the 32 k-rows:
            // global reads coalesce across lanes; LDS writes are 2x b128.
            int c  = tid & 127;          // 0..127
            int r0 = (tid >> 7) << 4;    // 0 or 16
            const float* src = Bm + (long)(k0 + r0) * N + (bn0 + c);
            v8h pk0, pk1;
            #pragma unroll
            for (int j = 0; j < 8; ++j)
                pk0[j] = (_Float16)src[(long)j * N];
            #pragma unroll
            for (int j = 0; j < 8; ++j)
                pk1[j] = (_Float16)src[(long)(8 + j) * N];
            *(v8h*)&Bst[c][r0]     = pk0;
            *(v8h*)&Bst[c][r0 + 8] = pk1;
        } else {
            for (int i = tid; i < BK * BN; i += 256) {
                int r = i >> 7, c = i & 127;
                int gr = k0 + r, gc = bn0 + c;
                Bst[c][r] = (_Float16)((gr < K && gc < N) ? Bm[(long)gr * N + gc] : 0.0f);
            }
        }

        // prefetch next K tile into caches while we compute this one
        if (k0 + BK < K) {
            int pr = bm0 + (tid & 127);
            if (pr < M) __builtin_prefetch(&A[(long)pr * K + k0 + BK], 0, 1);
            int br = k0 + BK + (tid >> 3);
            if (br < K) __builtin_prefetch(&Bm[(long)br * N + bn0 + ((tid & 7) << 4)], 0, 1);
        }
        __syncthreads();

        // ---------------- fragments: contiguous b128 LDS reads ----------------
        // A frag (16x32 f16, ISA 7.12.2): elems 0..7  -> As[row][half*8 + 0..7]
        //                                  elems 8..15 -> As[row][16 + half*8 + 0..7]
        // B frag: same k order as A -> Bst[col][half*8 + 0..7], Bst[col][16 + half*8 + 0..7].
        V16 af[2], bf[4];
        #pragma unroll
        for (int r = 0; r < 2; ++r) {
            const _Float16* row = &As[wm0 + r * 16 + l16][0];
            af[r].h[0] = *(const v8h*)(row + half * 8);
            af[r].h[1] = *(const v8h*)(row + 16 + half * 8);
        }
        #pragma unroll
        for (int c = 0; c < 4; ++c) {
            const _Float16* col = &Bst[wn0 + c * 16 + l16][0];
            bf[c].h[0] = *(const v8h*)(col + half * 8);
            bf[c].h[1] = *(const v8h*)(col + 16 + half * 8);
        }
        #pragma unroll
        for (int r = 0; r < 2; ++r)
            #pragma unroll
            for (int c = 0; c < 4; ++c)
                acc[r][c] = __builtin_amdgcn_wmma_f32_16x16x32_f16(
                    false, af[r].v, false, bf[c].v,
                    (short)0, acc[r][c], false, false);
        __syncthreads();
    }

    // ---------------- epilogue ----------------
    #pragma unroll
    for (int r = 0; r < 2; ++r)
        #pragma unroll
        for (int c = 0; c < 4; ++c)
            #pragma unroll
            for (int e = 0; e < 8; ++e) {
                int gr = bm0 + wm0 + r * 16 + half * 8 + e;
                int gc = bn0 + wn0 + c * 16 + l16;
                if (gr < M && gc < N) {
                    float x = acc[r][c][e];
                    if (bias) x += bias[gc];
                    if (res)  x += res[(long)gr * N + gc];
                    if (relu) x = fmaxf(x, 0.0f);
                    C[(long)gr * N + gc] = x;
                }
            }
}

// =====================================================================
// Dilated-conv branch: conv[b, p]  (p spans the 3 concatenated branches)
// =====================================================================
__global__ void conv_branch_kernel(
    const float* __restrict__ in0,
    const float* wd1, const float* bd1, const float* ws1, const float* bs1,
    const float* wd2, const float* bd2, const float* ws2, const float* bs2,
    const float* wd3, const float* bd3, const float* ws3, const float* bs3,
    float* __restrict__ conv)
{
    int idx = blockIdx.x * blockDim.x + threadIdx.x;
    if (idx >= B_ * INP_) return;
    int b = idx / INP_, p = idx % INP_;
    int d, t0;
    const float *wd, *bd, *ws, *bs;
    if (p < FLAT_ - 1)            { d = 1; t0 = p;                wd = wd1; bd = bd1; ws = ws1; bs = bs1; }
    else if (p < 2*FLAT_ - 3)     { d = 2; t0 = p - (FLAT_ - 1);  wd = wd2; bd = bd2; ws = ws2; bs = bs2; }
    else                          { d = 3; t0 = p - (2*FLAT_ - 3);wd = wd3; bd = bd3; ws = ws3; bs = bs3; }
    int L1 = FLAT_ - d;                     // per-channel conv length
    const float* f = in0 + (long)b * FLAT_;
    float acc = bs[0];
    #pragma unroll
    for (int k = 0; k < 2; ++k) {
        int j   = 2 * t0 + k;               // index into [NF*L1] channel-concat
        int ch  = j / L1;
        int pos = j % L1;
        float c = bd[ch] + f[pos] * wd[ch * 2 + 0] + f[pos + d] * wd[ch * 2 + 1];
        acc += c * ws[k];
    }
    conv[idx] = acc;
}

// =====================================================================
// GCN helpers
// =====================================================================
__global__ void deg_init_kernel(float* deg) {
    int n = blockIdx.x * blockDim.x + threadIdx.x;
    if (n < NN_) deg[n] = 0.0f;
}
__global__ void deg_acc_kernel(const int* __restrict__ ei, const float* __restrict__ ea,
                               float* __restrict__ deg) {
    int e = blockIdx.x * blockDim.x + threadIdx.x;
    if (e < E_)            atomicAdd(&deg[ei[E_ + e]], ea[e]);   // dst row
    else if (e < E_ + NN_) atomicAdd(&deg[e - E_], 1.0f);        // self loops
}
__global__ void norm_kernel(const int* __restrict__ ei, const float* __restrict__ ea,
                            const float* __restrict__ deg, float* __restrict__ nrm) {
    int e = blockIdx.x * blockDim.x + threadIdx.x;
    if (e >= E_ + NN_) return;
    int s, d; float w;
    if (e < E_) { s = ei[e]; d = ei[E_ + e]; w = ea[e]; }
    else        { s = d = e - E_;            w = 1.0f;  }
    float ds = deg[s] > 0.0f ? rsqrtf(deg[s]) : 0.0f;
    float dd = deg[d] > 0.0f ? rsqrtf(deg[d]) : 0.0f;
    nrm[e] = ds * w * dd;
}
// hw[b,n,o] = sum_s inputs[b,s,n] * w[s,o]   (layer 1, transposed read)
__global__ void gcn_mm1_kernel(const float* __restrict__ in0, const float* __restrict__ w,
                               float* __restrict__ hw) {
    int idx = blockIdx.x * blockDim.x + threadIdx.x;
    if (idx >= B_ * NN_ * H1_) return;
    int o = idx % H1_, n = (idx / H1_) % NN_, b = idx / (H1_ * NN_);
    float s = 0.0f;
    for (int ss = 0; ss < S_; ++ss)
        s += in0[((long)b * S_ + ss) * NN_ + n] * w[ss * H1_ + o];
    hw[idx] = s;
}
__global__ void gcn_mm2_kernel(const float* __restrict__ h, const float* __restrict__ w,
                               float* __restrict__ hw) {
    int idx = blockIdx.x * blockDim.x + threadIdx.x;
    if (idx >= B_ * NN_ * H2_) return;
    int o = idx % H2_, n = (idx / H2_) % NN_, b = idx / (H2_ * NN_);
    float s = 0.0f;
    for (int k = 0; k < H1_; ++k)
        s += h[((long)b * NN_ + n) * H1_ + k] * w[k * H2_ + o];
    hw[idx] = s;
}
__global__ void gcn_bias_init_kernel(const float* __restrict__ bias, float* __restrict__ out,
                                     int OUT) {
    int idx = blockIdx.x * blockDim.x + threadIdx.x;
    if (idx < B_ * NN_ * OUT) out[idx] = bias[idx % OUT];
}
__global__ void gcn_scatter_kernel(const float* __restrict__ hw, const int* __restrict__ ei,
                                   const float* __restrict__ nrm, float* __restrict__ out,
                                   int OUT) {
    int idx = blockIdx.x * blockDim.x + threadIdx.x;
    int b = blockIdx.y;
    if (idx >= (E_ + NN_) * OUT) return;
    int e = idx / OUT, o = idx % OUT;
    int s, d;
    if (e < E_) { s = ei[e]; d = ei[E_ + e]; } else { s = d = e - E_; }
    atomicAdd(&out[((long)b * NN_ + d) * OUT + o],
              hw[((long)b * NN_ + s) * OUT + o] * nrm[e]);
}
__global__ void relu_kernel(float* __restrict__ x, long n) {
    long idx = blockIdx.x * (long)blockDim.x + threadIdx.x;
    if (idx < n) x[idx] = fmaxf(x[idx], 0.0f);
}

// =====================================================================
// Fuse: t[b,n,:] = concat(v[b,:], h2[b,n,:])
// =====================================================================
__global__ void fuse_kernel(const float* __restrict__ v, const float* __restrict__ h2,
                            float* __restrict__ t) {
    long idx = blockIdx.x * (long)blockDim.x + threadIdx.x;
    if (idx >= (long)TOK_ * DD_) return;
    int c = (int)(idx % DD_);
    int tok = (int)(idx / DD_);
    int b = tok / NN_;
    t[idx] = (c < DV_) ? v[(long)b * DV_ + c] : h2[(long)tok * H2_ + (c - DV_)];
}

// =====================================================================
// Attention: one block per (query node, batch*head).
// =====================================================================
__global__ __launch_bounds__(256) void attention_kernel(const float* __restrict__ qkv,
                                                        float* __restrict__ out) {
    int q  = blockIdx.x;
    int bh = blockIdx.y;
    int b = bh / NH_, h = bh % NH_;
    __shared__ float qv[HD_];
    __shared__ float sc[NN_];
    __shared__ float inv_s;
    const float scale = rsqrtf((float)HD_);
    const float* qrow = qkv + (long)(b * NN_ + q) * (3 * DD_) + h * HD_;
    for (int d = threadIdx.x; d < HD_; d += blockDim.x) qv[d] = qrow[d];
    __syncthreads();
    for (int j = threadIdx.x; j < NN_; j += blockDim.x) {
        const float* krow = qkv + (long)(b * NN_ + j) * (3 * DD_) + DD_ + h * HD_;
        float s = 0.0f;
        for (int d = 0; d < HD_; ++d) s += qv[d] * krow[d];
        sc[j] = s * scale;
    }
    __syncthreads();
    if (threadIdx.x == 0) {
        float m = -1e30f;
        for (int j = 0; j < NN_; ++j) m = fmaxf(m, sc[j]);
        float sum = 0.0f;
        for (int j = 0; j < NN_; ++j) { float e = __expf(sc[j] - m); sc[j] = e; sum += e; }
        inv_s = 1.0f / sum;
    }
    __syncthreads();
    float inv = inv_s;
    for (int d = threadIdx.x; d < HD_; d += blockDim.x) {
        float o = 0.0f;
        for (int j = 0; j < NN_; ++j)
            o += sc[j] * qkv[(long)(b * NN_ + j) * (3 * DD_) + 2 * DD_ + h * HD_ + d];
        out[(long)(b * NN_ + q) * DD_ + h * HD_ + d] = o * inv;
    }
}

// =====================================================================
// LayerNorm over D per token, in place.
// =====================================================================
__global__ __launch_bounds__(256) void layernorm_kernel(float* __restrict__ t,
                                                        const float* __restrict__ g,
                                                        const float* __restrict__ be) {
    int tok = blockIdx.x;
    float* row = t + (long)tok * DD_;
    __shared__ float sbuf[256];
    float s = 0.0f;
    for (int i = threadIdx.x; i < DD_; i += 256) s += row[i];
    sbuf[threadIdx.x] = s; __syncthreads();
    for (int off = 128; off > 0; off >>= 1) {
        if (threadIdx.x < off) sbuf[threadIdx.x] += sbuf[threadIdx.x + off];
        __syncthreads();
    }
    float mean = sbuf[0] / DD_;
    __syncthreads();
    float v = 0.0f;
    for (int i = threadIdx.x; i < DD_; i += 256) { float d = row[i] - mean; v += d * d; }
    sbuf[threadIdx.x] = v; __syncthreads();
    for (int off = 128; off > 0; off >>= 1) {
        if (threadIdx.x < off) sbuf[threadIdx.x] += sbuf[threadIdx.x + off];
        __syncthreads();
    }
    float rstd = rsqrtf(sbuf[0] / DD_ + 1e-5f);
    __syncthreads();
    for (int i = threadIdx.x; i < DD_; i += 256)
        row[i] = (row[i] - mean) * rstd * g[i] + be[i];
}

// =====================================================================
// Host-side orchestration
// =====================================================================
static inline dim3 gemm_grid(int M, int N) { return dim3((N + BN - 1) / BN, (M + BM - 1) / BM); }

extern "C" void kernel_launch(void* const* d_in, const int* in_sizes, int n_in,
                              void* d_out, int out_size, void* d_ws, size_t ws_size,
                              hipStream_t stream) {
    const float* in0      = (const float*)d_in[0];
    const int*   eidx     = (const int*)  d_in[1];
    const float* eattr    = (const float*)d_in[2];
    const float* gcn1_w   = (const float*)d_in[3];
    const float* gcn1_b   = (const float*)d_in[4];
    const float* gcn2_w   = (const float*)d_in[5];
    const float* gcn2_b   = (const float*)d_in[6];
    const float* wd1 = (const float*)d_in[7],  *bd1 = (const float*)d_in[8];
    const float* ws1 = (const float*)d_in[9],  *bs1 = (const float*)d_in[10];
    const float* wd2 = (const float*)d_in[11], *bd2 = (const float*)d_in[12];
    const float* ws2 = (const float*)d_in[13], *bs2 = (const float*)d_in[14];
    const float* wd3 = (const float*)d_in[15], *bd3 = (const float*)d_in[16];
    const float* ws3 = (const float*)d_in[17], *bs3 = (const float*)d_in[18];
    const float* dense1_w = (const float*)d_in[19];
    const float* dense1_b = (const float*)d_in[20];
    const float* wqkv = (const float*)d_in[21], *bqkv = (const float*)d_in[22];
    const float* wo   = (const float*)d_in[23], *bo   = (const float*)d_in[24];
    const float* ln1g = (const float*)d_in[25], *ln1b = (const float*)d_in[26];
    const float* w1   = (const float*)d_in[27], *b1   = (const float*)d_in[28];
    const float* w2   = (const float*)d_in[29], *b2   = (const float*)d_in[30];
    const float* ln2g = (const float*)d_in[31], *ln2b = (const float*)d_in[32];
    const float* outw = (const float*)d_in[33], *outb = (const float*)d_in[34];
    float* out = (float*)d_out;

    // ---- workspace carve-out ----
    float* p = (float*)d_ws;
    float* conv = p; p += (long)B_ * INP_;
    float* v    = p; p += (long)B_ * DV_;
    float* deg  = p; p += NN_;
    float* nrm  = p; p += (E_ + NN_);
    float* hw1  = p; p += (long)B_ * NN_ * H1_;
    float* h1   = p; p += (long)B_ * NN_ * H1_;
    float* hw2  = p; p += (long)B_ * NN_ * H2_;
    float* h2   = p; p += (long)B_ * NN_ * H2_;
    float* t    = p; p += (long)TOK_ * DD_;
    float* qkv  = p; p += (long)TOK_ * 3 * DD_;
    float* attn = p; p += (long)TOK_ * DD_;
    float* ffn1 = p; p += (long)TOK_ * DFF_;

    // ---- conv branch + dense1 (WMMA) ----
    conv_branch_kernel<<<(B_ * INP_ + 255) / 256, 256, 0, stream>>>(
        in0, wd1, bd1, ws1, bs1, wd2, bd2, ws2, bs2, wd3, bd3, ws3, bs3, conv);
    gemm_wmma_f16<<<gemm_grid(B_, DV_), 256, 0, stream>>>(
        conv, dense1_w, dense1_b, nullptr, v, B_, DV_, INP_, 1);

    // ---- GCN branch ----
    deg_init_kernel<<<1, 128, 0, stream>>>(deg);
    deg_acc_kernel<<<(E_ + NN_ + 255) / 256, 256, 0, stream>>>(eidx, eattr, deg);
    norm_kernel<<<(E_ + NN_ + 255) / 256, 256, 0, stream>>>(eidx, eattr, deg, nrm);

    gcn_mm1_kernel<<<(B_ * NN_ * H1_ + 255) / 256, 256, 0, stream>>>(in0, gcn1_w, hw1);
    gcn_bias_init_kernel<<<(B_ * NN_ * H1_ + 255) / 256, 256, 0, stream>>>(gcn1_b, h1, H1_);
    gcn_scatter_kernel<<<dim3(((E_ + NN_) * H1_ + 255) / 256, B_), 256, 0, stream>>>(
        hw1, eidx, nrm, h1, H1_);
    relu_kernel<<<(B_ * NN_ * H1_ + 255) / 256, 256, 0, stream>>>(h1, (long)B_ * NN_ * H1_);

    gcn_mm2_kernel<<<(B_ * NN_ * H2_ + 255) / 256, 256, 0, stream>>>(h1, gcn2_w, hw2);
    gcn_bias_init_kernel<<<(B_ * NN_ * H2_ + 255) / 256, 256, 0, stream>>>(gcn2_b, h2, H2_);
    gcn_scatter_kernel<<<dim3(((E_ + NN_) * H2_ + 255) / 256, B_), 256, 0, stream>>>(
        hw2, eidx, nrm, h2, H2_);
    relu_kernel<<<(B_ * NN_ * H2_ + 255) / 256, 256, 0, stream>>>(h2, (long)B_ * NN_ * H2_);

    // ---- fuse ----
    fuse_kernel<<<(int)(((long)TOK_ * DD_ + 255) / 256), 256, 0, stream>>>(v, h2, t);

    // ---- transformer encoder (4 post-LN layers), all GEMMs on WMMA ----
    for (int i = 0; i < NL_; ++i) {
        gemm_wmma_f16<<<gemm_grid(TOK_, 3 * DD_), 256, 0, stream>>>(
            t, wqkv + (long)i * DD_ * 3 * DD_, bqkv + (long)i * 3 * DD_,
            nullptr, qkv, TOK_, 3 * DD_, DD_, 0);
        attention_kernel<<<dim3(NN_, B_ * NH_), 256, 0, stream>>>(qkv, attn);
        gemm_wmma_f16<<<gemm_grid(TOK_, DD_), 256, 0, stream>>>(
            attn, wo + (long)i * DD_ * DD_, bo + (long)i * DD_,
            t, t, TOK_, DD_, DD_, 0);                        // + residual, in place
        layernorm_kernel<<<TOK_, 256, 0, stream>>>(t, ln1g + (long)i * DD_, ln1b + (long)i * DD_);
        gemm_wmma_f16<<<gemm_grid(TOK_, DFF_), 256, 0, stream>>>(
            t, w1 + (long)i * DD_ * DFF_, b1 + (long)i * DFF_,
            nullptr, ffn1, TOK_, DFF_, DD_, 1);              // relu
        gemm_wmma_f16<<<gemm_grid(TOK_, DD_), 256, 0, stream>>>(
            ffn1, w2 + (long)i * DFF_ * DD_, b2 + (long)i * DD_,
            t, t, TOK_, DD_, DFF_, 0);                       // + residual, in place
        layernorm_kernel<<<TOK_, 256, 0, stream>>>(t, ln2g + (long)i * DD_, ln2b + (long)i * DD_);
    }

    // ---- output projection straight into d_out: [TOK, LBL] == [B, 1, N*LBL] ----
    gemm_wmma_f16<<<gemm_grid(TOK_, LBL_), 256, 0, stream>>>(
        t, outw, outb, nullptr, out, TOK_, LBL_, DD_, 0);
}
